// RelationNet_5669356833400
// MI455X (gfx1250) — matrix operand order repaired and checked by reading, not compiled, not run on hardware
//
#include <hip/hip_runtime.h>

#define DEV __device__ __forceinline__

typedef __attribute__((ext_vector_type(16))) __bf16   v16bf;
typedef __attribute__((ext_vector_type(8)))  float    v8f;
typedef __attribute__((ext_vector_type(4)))  unsigned v4u;
typedef __attribute__((ext_vector_type(8)))  unsigned v8u;

union BF16x16 { v16bf v; __bf16 e[16]; uint4 q[2]; };
union F32x8   { v8f   v; float  f[8];  };

static constexpr int Bn = 64, Nn = 100, Dn = 2048, NRn = 16, DKn = 64, DVn = 128;
static constexpr int Pn = Bn * Nn;   // 6400 rows

// ---- WMMA fragment index maps (CDNA5 ISA 7.12.2, 16-bit data, wave32) ----
// A 16x32: lane group g, M=lane&15; e<8 -> K=e+8g ; e>=8 -> K=e+8+8g
//   => two contiguous 8-element runs at K = 8g and K = 16+8g
// B 32x16: N=lane&15; K(e) same as A => two 8-element runs at K = 8g and K = 16+8g
// C/D 16x16 f32: VGPR r -> M = r + 8g, N = lane&15
DEV int kidxA(int g, int e) { return (e < 8) ? (e + 8 * g) : (e + 8 + 8 * g); }

DEV v8f wmma_bf16(v16bf a, v16bf b, v8f c) {
  return __builtin_amdgcn_wmma_f32_16x16x32_bf16(false, a, false, b, (short)0, c,
                                                 false, false);
}
// load A fragment: base = row pointer, K-chunks at kb+8g and kb+16+8g
DEV void ldA(BF16x16& fr, const __bf16* rowp, int kb, int gh) {
  fr.q[0] = *(const uint4*)(rowp + kb + 8 * gh);
  fr.q[1] = *(const uint4*)(rowp + kb + 16 + 8 * gh);
}
// load B fragment from an [N][K] LDS layout: K-chunks at kb+8g and kb+16+8g
DEV void ldB(BF16x16& fr, const __bf16* rowp, int kb, int gh) {
  fr.q[0] = *(const uint4*)(rowp + kb + 8 * gh);
  fr.q[1] = *(const uint4*)(rowp + kb + 16 + 8 * gh);
}

// ---- Tensor Data Mover: 2D tile (global -> LDS), D# packed per ISA 8.3/8.4 ----
DEV void tdm_load_2d(unsigned lds_byte_off, const void* global_ptr,
                     unsigned tile_d0, unsigned tile_d1,
                     unsigned tensor_d0, unsigned tensor_d1,
                     unsigned long long row_stride_elems,
                     unsigned data_size_code /*0:1B 1:2B 2:4B 3:8B*/) {
  unsigned long long ga = (unsigned long long)global_ptr;
  v4u g0;
  g0[0] = 1u;                                                   // count=1, no gather
  g0[1] = lds_byte_off;                                         // lds_addr
  g0[2] = (unsigned)ga;                                         // global_addr[31:0]
  g0[3] = (unsigned)((ga >> 32) & 0x01FFFFFFull) | 0x80000000u; // ga[56:32] | type=2
  v8u g1;
  g1[0] = data_size_code << 16;                                 // wg_mask=0, data_size
  g1[1] = (tensor_d0 & 0xFFFFu) << 16;                          // tensor_dim0 lo
  g1[2] = (tensor_d0 >> 16) | ((tensor_d1 & 0xFFFFu) << 16);    // dim0 hi | dim1 lo
  g1[3] = (tensor_d1 >> 16) | (tile_d0 << 16);                  // dim1 hi | tile_dim0
  g1[4] = tile_d1 & 0xFFFFu;                                    // tile_dim1, tile_dim2=0
  g1[5] = (unsigned)row_stride_elems;                           // dim0_stride lo
  g1[6] = (unsigned)((row_stride_elems >> 32) & 0xFFFFull);     // dim0_stride hi
  g1[7] = 0u;
  asm volatile("tensor_load_to_lds %0, %1" :: "s"(g0), "s"(g1) : "memory");
}

// =============== K0: precision convert / weight pack (W stored transposed) =====
__global__ void k_convert(const float* __restrict__ a, const float* __restrict__ wk,
                          const float* __restrict__ wq, const float* __restrict__ bk,
                          const float* __restrict__ bq, __bf16* __restrict__ abf,
                          __bf16* __restrict__ wkqT, float* __restrict__ bkq) {
  long long tid = (long long)blockIdx.x * blockDim.x + threadIdx.x;
  long long stride = (long long)gridDim.x * blockDim.x;
  const long long nA = (long long)Pn * Dn;
  for (long long i = tid; i < nA; i += stride) abf[i] = (__bf16)a[i];
  const long long nW = (long long)Dn * Dn;
  for (long long i = tid; i < nW; i += stride) {   // i = c*2048 + k  (transposed)
    int k = (int)(i & (Dn - 1));
    long long c = i >> 11;
    float v = (c < 1024) ? wk[(long long)k * 1024 + c]
                         : wq[(long long)k * 1024 + (c - 1024)];
    wkqT[i] = (__bf16)v;
  }
  for (long long i = tid; i < Dn; i += stride)
    bkq[i] = (i < 1024) ? bk[i] : bq[i - 1024];
}

// =============== K1: gate = log(max(g@Wg+bg,1e-6)), WMMA skinny GEMM ===========
__global__ void k_gate(const float* __restrict__ g, const float* __restrict__ Wg,
                       const float* __restrict__ bg, float* __restrict__ lgw) {
  __shared__ __bf16 sWgT[16][64];    // transposed: [h][k] -> B frags contiguous
  int tid = threadIdx.x;
  for (int i = tid; i < 64 * 16; i += blockDim.x) {
    int k = i >> 4, h = i & 15;
    sWgT[h][k] = (__bf16)Wg[i];
  }
  __syncthreads();
  int lane = tid & 31, gh = lane >> 4, nl = lane & 15, wid = tid >> 5;

  BF16x16 bf0, bf1;                  // B fragments (Wg) for k-base 0 and 32
  ldB(bf0, &sWgT[nl][0], 0, gh);
  ldB(bf1, &sWgT[nl][0], 32, gh);

  const int NGROUP = (Bn * Nn * Nn) / 16;  // 40000 row-groups of 16
  int gwave = blockIdx.x * (blockDim.x >> 5) + wid;
  int nw = gridDim.x * (blockDim.x >> 5);
  for (int grp = gwave; grp < NGROUP; grp += nw) {
    long long rowbase = (long long)grp * 16;
    const float* rp = g + (rowbase + nl) * 64;   // compiler emits global_load_b128
    BF16x16 a0, a1;
#pragma unroll
    for (int e = 0; e < 16; e++) {
      a0.e[e] = (__bf16)rp[kidxA(gh, e)];
      a1.e[e] = (__bf16)rp[32 + kidxA(gh, e)];
    }
    F32x8 acc;
#pragma unroll
    for (int r = 0; r < 8; r++) acc.f[r] = 0.f;
    acc.v = wmma_bf16(a0.v, bf0.v, acc.v);
    acc.v = wmma_bf16(a1.v, bf1.v, acc.v);
#pragma unroll
    for (int r = 0; r < 8; r++) {
      long long row = rowbase + r + 8 * gh;
      float val = fmaxf(acc.f[r] + bg[nl], 1e-6f);
      lgw[row * NRn + nl] = __logf(val);
    }
  }
}

// ====== K2: [6400x2048]x[2048x2048] bf16 GEMM (K|Q fused), TDM double-buffered ====
__global__ void k_gemm_kq(const __bf16* __restrict__ abf, const __bf16* __restrict__ wkqT,
                          const float* __restrict__ bkq, __bf16* __restrict__ kqbf) {
  __shared__ __bf16 sA [2][128][32];   // [m][k] 16 KB
  __shared__ __bf16 sBt[2][128][32];   // [n][k] 16 KB  (weight pre-transposed)
  const int mbase = blockIdx.x * 128, nbase = blockIdx.y * 128;
  int tid = threadIdx.x;
  int wid = tid >> 5, lane = tid & 31, gh = lane >> 4, nl = lane & 15;
  const int NK = Dn / 32;  // 64 k-steps

  if (wid == 0) {  // prologue: fill buffer 0
    tdm_load_2d((unsigned)(unsigned long long)&sA[0][0][0],
                abf + (long long)mbase * Dn, 32, 128, Dn, Pn, Dn, 1);
    tdm_load_2d((unsigned)(unsigned long long)&sBt[0][0][0],
                wkqT + (long long)nbase * Dn, 32, 128, Dn, Dn, Dn, 1);
  }

  F32x8 acc[8];
  for (int nt = 0; nt < 8; nt++)
    for (int r = 0; r < 8; r++) acc[nt].f[r] = 0.f;

  for (int k = 0; k < NK; k++) {
    int buf = k & 1;
    if (wid == 0) {
      if (k + 1 < NK) {  // prefetch next tile, wait only for the current one
        int k1 = (k + 1) * 32;
        tdm_load_2d((unsigned)(unsigned long long)&sA[buf ^ 1][0][0],
                    abf + (long long)mbase * Dn + k1, 32, 128, Dn, Pn, Dn, 1);
        tdm_load_2d((unsigned)(unsigned long long)&sBt[buf ^ 1][0][0],
                    wkqT + (long long)nbase * Dn + k1, 32, 128, Dn, Dn, Dn, 1);
        __builtin_amdgcn_s_wait_tensorcnt(2);
      } else {
        __builtin_amdgcn_s_wait_tensorcnt(0);
      }
    }
    __syncthreads();
    BF16x16 af;  // wave wid owns M-strip [16*wid, 16*wid+16)
    ldA(af, &sA[buf][16 * wid + nl][0], 0, gh);
#pragma unroll
    for (int nt = 0; nt < 8; nt++) {
      BF16x16 bfg;
      ldB(bfg, &sBt[buf][nt * 16 + nl][0], 0, gh);
      acc[nt].v = wmma_bf16(af.v, bfg.v, acc[nt].v);
    }
    __syncthreads();
  }

#pragma unroll
  for (int nt = 0; nt < 8; nt++)
#pragma unroll
    for (int r = 0; r < 8; r++) {
      int row = mbase + 16 * wid + r + 8 * gh;
      int col = nbase + nt * 16 + nl;
      kqbf[(long long)row * Dn + col] = (__bf16)(acc[nt].f[r] + bkq[col]);
    }
}

// ====== K3: per-(b,h) attention: S=KQ^T/8+lgw, softmax over i, R=P^T V ======
__global__ void k_attn(const __bf16* __restrict__ kqbf, const __bf16* __restrict__ abf,
                       const float* __restrict__ lgw, const float* __restrict__ wsp,
                       const float* __restrict__ bsp, float* __restrict__ out) {
  // Hand-placed LDS arena with phase-accurate reuse (132 KB):
  //  [0,32K)      : K[128][64], Q[128][64]      -> later Pt[128][136]
  //  [34816,69632): Vt[128][136] (transposed V)
  //  [69632,end)  : V staging bf16[128][128]    -> later S f32[128][128]
  __shared__ __align__(16) char smem[135168];
  __bf16* sK  = (__bf16*)smem;                 // [128][64]
  __bf16* sQ  = (__bf16*)(smem + 16384);       // [128][64]
  __bf16* Pt  = (__bf16*)smem;                 // [128][136], P^T
  __bf16* sVt = (__bf16*)(smem + 34816);       // [128][136], V^T
  __bf16* Vs  = (__bf16*)(smem + 69632);       // staging [128][128]
  float*  sS  = (float*) (smem + 69632);       // [128][128]

  const int b = blockIdx.x >> 4, h = blockIdx.x & 15;
  int tid = threadIdx.x;
  int wid = tid >> 5, lane = tid & 31, gh = lane >> 4, nl = lane & 15;

  if (wid == 0) {  // TDM: tensor_dim1=100 -> rows >= 100 auto zero-filled
    const __bf16* kp = kqbf + (long long)b * Nn * Dn + h * DKn;
    const __bf16* qp = kp + 1024;
    const __bf16* vp = abf + (long long)b * Nn * Dn + h * DVn;
    tdm_load_2d((unsigned)(unsigned long long)sK, kp, 64, 128, 64, Nn, Dn, 1);
    tdm_load_2d((unsigned)(unsigned long long)sQ, qp, 64, 128, 64, Nn, Dn, 1);
    tdm_load_2d((unsigned)(unsigned long long)Vs, vp, 128, 128, 128, Nn, Dn, 1);
    __builtin_amdgcn_s_wait_tensorcnt(0);
  }
  __syncthreads();

  // Transpose V staging -> Vt[d][i] so phase-C B fragments are contiguous
  for (int idx = tid; idx < 128 * 128; idx += 256) {
    int i = idx >> 7, d = idx & 127;
    sVt[d * 136 + i] = Vs[idx];
  }
  __syncthreads();

  {  // Phase A: S[i][j] = (K_i . Q_j)/8 + lgw ; wave wid -> i-strip [16*wid,+16)
    F32x8 acc[8];
    for (int jt = 0; jt < 8; jt++)
      for (int r = 0; r < 8; r++) acc[jt].f[r] = 0.f;
#pragma unroll
    for (int ks = 0; ks < 2; ks++) {
      int kb = ks * 32;
      BF16x16 af;
      ldA(af, sK + (16 * wid + nl) * 64, kb, gh);
#pragma unroll
      for (int jt = 0; jt < 8; jt++) {
        BF16x16 bfg;  // Q is [j][d] == [N][K] layout: contiguous
        ldB(bfg, sQ + (jt * 16 + nl) * 64, kb, gh);
        acc[jt].v = wmma_bf16(af.v, bfg.v, acc[jt].v);
      }
    }
#pragma unroll
    for (int jt = 0; jt < 8; jt++)
#pragma unroll
      for (int r = 0; r < 8; r++) {
        int i = 16 * wid + r + 8 * gh, j = jt * 16 + nl;
        float v = -1e30f;  // padded key rows vanish under softmax
        if (i < Nn && j < Nn) {
          long long p = ((long long)b * Nn + i) * Nn + j;
          v = acc[jt].f[r] * 0.125f + lgw[p * NRn + h];
        }
        sS[i * 128 + j] = v;
      }
  }
  __syncthreads();

  // Phase B: softmax over i per column j; write P^T (bf16) over the K/Q space
  if (tid < 128) {
    int j = tid;
    float m = -1e30f;
    for (int i = 0; i < 128; i++) m = fmaxf(m, sS[i * 128 + j]);
    float s = 0.f;
    for (int i = 0; i < 128; i++) {
      float e = __expf(sS[i * 128 + j] - m);
      sS[i * 128 + j] = e;
      s += e;
    }
    float rinv = 1.f / s;
    for (int i = 0; i < 128; i++) Pt[j * 136 + i] = (__bf16)(sS[i * 128 + j] * rinv);
  }
  __syncthreads();

  {  // Phase C: R[j][d] = sum_i P[i][j] V[i][d]; A = P^T (from Pt), B = V (from Vt)
    F32x8 acc[8];
    for (int dt = 0; dt < 8; dt++)
      for (int r = 0; r < 8; r++) acc[dt].f[r] = 0.f;
#pragma unroll
    for (int ks = 0; ks < 4; ks++) {
      int kb = ks * 32;
      BF16x16 af;
      ldA(af, Pt + (16 * wid + nl) * 136, kb, gh);
#pragma unroll
      for (int dt = 0; dt < 8; dt++) {
        BF16x16 bfg;
        ldB(bfg, sVt + (dt * 16 + nl) * 136, kb, gh);
        acc[dt].v = wmma_bf16(af.v, bfg.v, acc[dt].v);
      }
    }
    float wsv = *wsp, bsv = *bsp;
#pragma unroll
    for (int dt = 0; dt < 8; dt++)
#pragma unroll
      for (int r = 0; r < 8; r++) {
        int j = 16 * wid + r + 8 * gh, d = dt * 16 + nl;
        if (j < Nn)
          out[((long long)b * Nn + j) * Dn + h * DVn + d] = acc[dt].f[r] * wsv + bsv;
      }
  }
}

extern "C" void kernel_launch(void* const* d_in, const int* in_sizes, int n_in,
                              void* d_out, int out_size, void* d_ws, size_t ws_size,
                              hipStream_t stream) {
  const float* a  = (const float*)d_in[0];
  const float* g  = (const float*)d_in[1];
  // d_in[2] split, d_in[3] rel_pair_counts: uniform splits, unused
  const float* Wg = (const float*)d_in[4];
  const float* bg = (const float*)d_in[5];
  const float* Wk = (const float*)d_in[6];
  const float* bk = (const float*)d_in[7];
  const float* Wq = (const float*)d_in[8];
  const float* bq = (const float*)d_in[9];
  const float* ws = (const float*)d_in[10];
  const float* bs = (const float*)d_in[11];
  float* out = (float*)d_out;

  char* p = (char*)d_ws;
  auto alloc = [&](size_t bytes) { char* r = p; p += (bytes + 255) & ~(size_t)255; return r; };
  __bf16* abf  = (__bf16*)alloc((size_t)Pn * Dn * 2);            // a in bf16
  __bf16* wkqT = (__bf16*)alloc((size_t)Dn * Dn * 2);            // [Wk|Wq]^T bf16
  float*  bkq  = (float*) alloc((size_t)Dn * 4);                 // [bk|bq]
  __bf16* kqbf = (__bf16*)alloc((size_t)Pn * Dn * 2);            // K|Q bf16
  float*  lgw  = (float*) alloc((size_t)Bn * Nn * Nn * NRn * 4); // log-gate

  k_convert<<<2048, 256, 0, stream>>>(a, Wk, Wq, bk, bq, abf, wkqT, bkq);
  k_gate<<<1024, 256, 0, stream>>>(g, Wg, bg, lgw);
  k_gemm_kq<<<dim3(Pn / 128, Dn / 128), 256, 0, stream>>>(abf, wkqT, bkq, kqbf);
  k_attn<<<Bn * NRn, 256, 0, stream>>>(kqbf, abf, lgw, ws, bs, out);
}
